// VllmMixtureOfExpertsOpWNA16_71141838291321
// MI455X (gfx1250) — matrix operand order repaired and checked by reading, not compiled, not run on hardware
//
#include <hip/hip_runtime.h>
#include <hip/hip_bf16.h>

// ---------------- problem constants (match reference setup_inputs) ----------
#define T_TOK   2048
#define DDIM    2048
#define IDIM    2816
#define NEXP    8
#define QGROUP  128
#define TOPK    2
#define MTILE   64
#define NTILE   64
#define KCHUNK  64
#define RTILES  72                        // >= max sum of ceil(cnt_e/64) = 71
#define MAXSLOTS (RTILES * MTILE)         // 4608

typedef __attribute__((ext_vector_type(16))) __bf16 v16bf;
typedef __attribute__((ext_vector_type(8)))  float  v8f;

// ---- gfx1250 async copy: global -> LDS, per-lane 16B, tracked by ASYNCcnt --
__device__ __forceinline__ void async_ld_b128(void* lds_ptr, const void* gptr) {
    unsigned lds_off = (unsigned)(unsigned long long)lds_ptr;  // low 32b = LDS addr
    unsigned long long ga = (unsigned long long)gptr;
    asm volatile("global_load_async_to_lds_b128 %0, %1, off"
                 :: "v"(lds_off), "v"(ga) : "memory");
}
__device__ __forceinline__ void wait_async0() {
    asm volatile("s_wait_asynccnt 0x0" ::: "memory");
}

// ---------------------------------------------------------------------------
// Routing / setup kernels
// ---------------------------------------------------------------------------
__global__ void zero_out_kernel(float* __restrict__ out, long long n) {
    long long i = (long long)blockIdx.x * blockDim.x + threadIdx.x;
    if (i < n) out[i] = 0.0f;
}

__global__ void zero_meta_kernel(int* counts, int* cursor, int* tile_expert,
                                 int* slot_token) {
    int i = blockIdx.x * blockDim.x + threadIdx.x;
    if (i < NEXP) { counts[i] = 0; cursor[i] = 0; }
    if (i < 96) tile_expert[i] = -1;
    if (i < MAXSLOTS) slot_token[i] = -1;
}

__global__ void count_kernel(const int* __restrict__ topk_ids, int* counts) {
    int i = blockIdx.x * blockDim.x + threadIdx.x;
    if (i < T_TOK * TOPK) atomicAdd(&counts[topk_ids[i]], 1);
}

__global__ void build_meta_kernel(const int* __restrict__ counts, int* offs,
                                  int* tile_expert, int* tile_slot0) {
    if (threadIdx.x != 0 || blockIdx.x != 0) return;
    int off = 0, tidx = 0;
    for (int e = 0; e < NEXP; ++e) {
        offs[e] = off;
        int cnt = counts[e];
        int ntile = (cnt + MTILE - 1) / MTILE;
        for (int t = 0; t < ntile; ++t) {
            tile_expert[tidx] = e;
            tile_slot0[tidx] = off + t * MTILE;
            ++tidx;
        }
        off += ntile * MTILE;
    }
    offs[NEXP] = off;
}

__global__ void scatter_kernel(const int* __restrict__ topk_ids,
                               const float* __restrict__ topk_w,
                               const int* __restrict__ offs, int* cursor,
                               int* slot_token, float* slot_weight) {
    int i = blockIdx.x * blockDim.x + threadIdx.x;
    if (i >= T_TOK * TOPK) return;
    int e = topk_ids[i];
    int slot = offs[e] + atomicAdd(&cursor[e], 1);
    slot_token[slot] = i / TOPK;
    slot_weight[slot] = topk_w[i];
}

__global__ void convert_x_kernel(const float* __restrict__ x,
                                 __bf16* __restrict__ xb, long long n) {
    long long i = (long long)blockIdx.x * blockDim.x + threadIdx.x;
    if (i < n) xb[i] = (__bf16)x[i];
}

// ---------------------------------------------------------------------------
// GEMM1: h[slot, n] = silu(x . W13g[n]) * (x . W13u[n]),  bf16 WMMA
// block: 256 thr (8 wave32). M=64 slots x N=64 columns of I. K over D.
// Waves tiled 2(mi) x 4(ni); each wave owns two 16-row subtiles (B reuse x2).
// ---------------------------------------------------------------------------
__global__ __launch_bounds__(256) void gemm1_kernel(
    const __bf16* __restrict__ xb,       // [T, D] bf16
    const int*    __restrict__ w13p,     // [E, 2I, D/2]
    const float*  __restrict__ w13s,     // [E, 2I, D/QGROUP]
    const float*  __restrict__ w13z,
    const int*    __restrict__ tile_expert,
    const int*    __restrict__ tile_slot0,
    const int*    __restrict__ slot_token,
    __bf16*       __restrict__ hbuf)     // [MAXSLOTS, I] bf16
{
    const int r = blockIdx.x;
    const int e = tile_expert[r];
    if (e < 0) return;
    const int slot0 = tile_slot0[r];
    const int nb = blockIdx.y * NTILE;   // gate-neuron base within I

    __shared__ __bf16 lA [MTILE][KCHUNK + 8];
    __shared__ __bf16 lBg[NTILE][KCHUNK + 8];  // [neuron][k]
    __shared__ __bf16 lBu[NTILE][KCHUNK + 8];

    const int tid  = threadIdx.x;
    const int lane = tid & 31;
    const int wave = tid >> 5;
    const int mi = wave >> 2, ni = wave & 3;   // 2x4 wave tiling
    const int lrow = lane & 15, lhi = lane >> 4;

    v8f accG0 = {}, accG1 = {}, accU0 = {}, accU1 = {};

    const int a_r0 = tid >> 3;           // 0..31
    const int a_r1 = a_r0 + 32;          // 32..63
    const int a_kk = (tid & 7) * 8;      // 0..56
    const int a_tok0 = slot_token[slot0 + a_r0];
    const int a_tok1 = slot_token[slot0 + a_r1];

    const long long wrow = (long long)(DDIM / 2);
    const long long srow = (long long)(DDIM / QGROUP);
    const long long baseG = (long long)e * (2 * IDIM) + nb;
    const long long baseU = baseG + IDIM;

    for (int k0 = 0; k0 < DDIM; k0 += KCHUNK) {
        __syncthreads();
        // ---- async-stage A (x tile), overlaps with weight dequant below ----
        if (a_tok0 >= 0)
            async_ld_b128(&lA[a_r0][a_kk], xb + (long long)a_tok0 * DDIM + k0 + a_kk);
        else {
            #pragma unroll
            for (int j = 0; j < 8; ++j) lA[a_r0][a_kk + j] = (__bf16)0.0f;
        }
        if (a_tok1 >= 0)
            async_ld_b128(&lA[a_r1][a_kk], xb + (long long)a_tok1 * DDIM + k0 + a_kk);
        else {
            #pragma unroll
            for (int j = 0; j < 8; ++j) lA[a_r1][a_kk + j] = (__bf16)0.0f;
        }
        // ---- dequant W13 gate+up chunk into LDS (VALU, overlapped w/ DMA) --
        {
            const int gidx = k0 / QGROUP;   // one scale/zp per neuron per chunk
            #pragma unroll
            for (int it = 0; it < 8; ++it) {
                int idx = tid * 8 + it;          // 0..2047
                int nl = idx >> 5;               // local neuron 0..63
                int p  = idx & 31;               // packed int32 index
                {
                    long long nr = baseG + nl;
                    int v   = w13p[nr * wrow + (k0 >> 1) + p];
                    float s = w13s[nr * srow + gidx];
                    float z = w13z[nr * srow + gidx];
                    lBg[nl][2 * p]     = (__bf16)(((float)(v & 15) - z) * s);
                    lBg[nl][2 * p + 1] = (__bf16)(((float)((v >> 4) & 15) - z) * s);
                }
                {
                    long long nr = baseU + nl;
                    int v   = w13p[nr * wrow + (k0 >> 1) + p];
                    float s = w13s[nr * srow + gidx];
                    float z = w13z[nr * srow + gidx];
                    lBu[nl][2 * p]     = (__bf16)(((float)(v & 15) - z) * s);
                    lBu[nl][2 * p + 1] = (__bf16)(((float)((v >> 4) & 15) - z) * s);
                }
            }
            if (k0 + KCHUNK < DDIM)  // pull next weight chunk toward L2/L0
                __builtin_prefetch(&w13p[(baseG + (tid >> 2)) * wrow +
                                         ((k0 + KCHUNK) >> 1) + (tid & 3) * 8], 0, 1);
        }
        wait_async0();
        __syncthreads();
        // ---- two 16x16x32 WMMA K-substeps per chunk, 8 WMMAs per wave ------
        #pragma unroll
        for (int ks = 0; ks < KCHUNK; ks += 32) {
            v16bf af0, af1, bg, bu;
            const int ar0 = mi * 32 + lrow;
            const int ar1 = ar0 + 16;
            const int ab  = ks + lhi * 16;       // A: K 0-15 vs 16-31
            #pragma unroll
            for (int j = 0; j < 8; ++j) {
                af0[j]     = lA[ar0][ab + j];
                af0[8 + j] = lA[ar0][ab + 8 + j];
                af1[j]     = lA[ar1][ab + j];
                af1[8 + j] = lA[ar1][ab + 8 + j];
            }
            const int bcol = ni * 16 + lrow;     // B: lane = column
            const int bb   = ks + lhi * 16;      // K-half by lane-half
            #pragma unroll
            for (int j = 0; j < 16; ++j) {
                bg[j] = lBg[bcol][bb + j];
                bu[j] = lBu[bcol][bb + j];
            }
            accG0 = __builtin_amdgcn_wmma_f32_16x16x32_bf16(
                false, af0, false, bg, (short)0, accG0, false, false);
            accG1 = __builtin_amdgcn_wmma_f32_16x16x32_bf16(
                false, af1, false, bg, (short)0, accG1, false, false);
            accU0 = __builtin_amdgcn_wmma_f32_16x16x32_bf16(
                false, af0, false, bu, (short)0, accU0, false, false);
            accU1 = __builtin_amdgcn_wmma_f32_16x16x32_bf16(
                false, af1, false, bu, (short)0, accU1, false, false);
        }
    }
    // ---- h = silu(g) * u ----
    const int col = nb + ni * 16 + lrow;
    #pragma unroll
    for (int j = 0; j < 8; ++j) {
        int m0 = mi * 32 + lhi * 8 + j;          // D-layout: VGPR j -> M=j / j+8
        float g0 = accG0[j], u0 = accU0[j];
        hbuf[(long long)(slot0 + m0) * IDIM + col] =
            (__bf16)((g0 / (1.0f + __expf(-g0))) * u0);
        float g1 = accG1[j], u1 = accU1[j];
        hbuf[(long long)(slot0 + m0 + 16) * IDIM + col] =
            (__bf16)((g1 / (1.0f + __expf(-g1))) * u1);
    }
}

// ---------------------------------------------------------------------------
// GEMM2: out[token, d] += route_w * (h[slot, :] . W2[e, d, :])
// ---------------------------------------------------------------------------
__global__ __launch_bounds__(256) void gemm2_kernel(
    const __bf16* __restrict__ hbuf,     // [MAXSLOTS, I]
    const int*    __restrict__ w2p,      // [E, D, I/2]
    const float*  __restrict__ w2s,      // [E, D, I/QGROUP]
    const float*  __restrict__ w2z,
    const int*    __restrict__ tile_expert,
    const int*    __restrict__ tile_slot0,
    const int*    __restrict__ slot_token,
    const float*  __restrict__ slot_weight,
    float*        __restrict__ out)      // [T, D] fp32
{
    const int r = blockIdx.x;
    const int e = tile_expert[r];
    if (e < 0) return;
    const int slot0 = tile_slot0[r];
    const int db = blockIdx.y * NTILE;   // output-d base

    __shared__ __bf16 lA[MTILE][KCHUNK + 8];
    __shared__ __bf16 lB[NTILE][KCHUNK + 8];

    const int tid  = threadIdx.x;
    const int lane = tid & 31;
    const int wave = tid >> 5;
    const int mi = wave >> 2, ni = wave & 3;
    const int lrow = lane & 15, lhi = lane >> 4;

    v8f acc0 = {}, acc1 = {};

    const int a_r0 = tid >> 3;
    const int a_r1 = a_r0 + 32;
    const int a_kk = (tid & 7) * 8;

    const long long wrow = (long long)(IDIM / 2);
    const long long srow = (long long)(IDIM / QGROUP);
    const long long baseD = (long long)e * DDIM + db;

    for (int k0 = 0; k0 < IDIM; k0 += KCHUNK) {
        __syncthreads();
        // padded slots hold zeros written by gemm1, so load unconditionally
        async_ld_b128(&lA[a_r0][a_kk],
                      hbuf + (long long)(slot0 + a_r0) * IDIM + k0 + a_kk);
        async_ld_b128(&lA[a_r1][a_kk],
                      hbuf + (long long)(slot0 + a_r1) * IDIM + k0 + a_kk);
        {   // dequant W2 chunk: 64 d-rows x 32 packed each; 256 thr * 8
            const int gidx = k0 / QGROUP;
            #pragma unroll
            for (int it = 0; it < 8; ++it) {
                int idx = tid * 8 + it;
                int nl = idx >> 5;
                int p  = idx & 31;
                long long nr = baseD + nl;
                int v   = w2p[nr * wrow + (k0 >> 1) + p];
                float s = w2s[nr * srow + gidx];
                float z = w2z[nr * srow + gidx];
                lB[nl][2 * p]     = (__bf16)(((float)(v & 15) - z) * s);
                lB[nl][2 * p + 1] = (__bf16)(((float)((v >> 4) & 15) - z) * s);
            }
            if (k0 + KCHUNK < IDIM)
                __builtin_prefetch(&w2p[(baseD + (tid >> 2)) * wrow +
                                        ((k0 + KCHUNK) >> 1) + (tid & 3) * 8], 0, 1);
        }
        wait_async0();
        __syncthreads();
        #pragma unroll
        for (int ks = 0; ks < KCHUNK; ks += 32) {
            v16bf af0, af1, bf;
            const int ar0 = mi * 32 + lrow;
            const int ar1 = ar0 + 16;
            const int ab  = ks + lhi * 16;
            #pragma unroll
            for (int j = 0; j < 8; ++j) {
                af0[j]     = lA[ar0][ab + j];
                af0[8 + j] = lA[ar0][ab + 8 + j];
                af1[j]     = lA[ar1][ab + j];
                af1[8 + j] = lA[ar1][ab + 8 + j];
            }
            const int bcol = ni * 16 + lrow;
            const int bb   = ks + lhi * 16;
            #pragma unroll
            for (int j = 0; j < 16; ++j) bf[j] = lB[bcol][bb + j];
            acc0 = __builtin_amdgcn_wmma_f32_16x16x32_bf16(
                false, af0, false, bf, (short)0, acc0, false, false);
            acc1 = __builtin_amdgcn_wmma_f32_16x16x32_bf16(
                false, af1, false, bf, (short)0, acc1, false, false);
        }
    }
    const int col = db + ni * 16 + lrow;
    #pragma unroll
    for (int j = 0; j < 8; ++j) {
        int m0 = mi * 32 + lhi * 8 + j;
        int s0 = slot0 + m0, s1 = s0 + 16;
        int t0 = slot_token[s0], t1 = slot_token[s1];
        if (t0 >= 0)
            atomicAdd(&out[(long long)t0 * DDIM + col], slot_weight[s0] * acc0[j]);
        if (t1 >= 0)
            atomicAdd(&out[(long long)t1 * DDIM + col], slot_weight[s1] * acc1[j]);
    }
}

// ---------------------------------------------------------------------------
extern "C" void kernel_launch(void* const* d_in, const int* in_sizes, int n_in,
                              void* d_out, int out_size, void* d_ws, size_t ws_size,
                              hipStream_t stream) {
    const float* x        = (const float*)d_in[0];
    const int*   topk_ids = (const int*)d_in[1];
    const float* topk_w   = (const float*)d_in[2];
    const int*   w13p     = (const int*)d_in[3];
    const float* w13s     = (const float*)d_in[4];
    const float* w13z     = (const float*)d_in[5];
    const int*   w2p      = (const int*)d_in[6];
    const float* w2s      = (const float*)d_in[7];
    const float* w2z      = (const float*)d_in[8];
    float* out = (float*)d_out;

    // ---- workspace carving ----
    int* counts      = (int*)d_ws;                 // 8
    int* cursor      = counts + 8;                 // 8
    int* offs        = cursor + 8;                 // 16 (9 used, padded)
    int* tile_expert = offs + 16;                  // 96
    int* tile_slot0  = tile_expert + 96;           // 96
    int* slot_token  = tile_slot0 + 96;            // MAXSLOTS
    float* slot_weight = (float*)(slot_token + MAXSLOTS);   // MAXSLOTS
    __bf16* xb   = (__bf16*)(slot_weight + MAXSLOTS);       // T*D
    __bf16* hbuf = xb + (long long)T_TOK * DDIM;            // MAXSLOTS*I

    const long long nout = (long long)T_TOK * DDIM;
    zero_out_kernel<<<(unsigned)((nout + 255) / 256), 256, 0, stream>>>(out, nout);
    zero_meta_kernel<<<(MAXSLOTS + 255) / 256, 256, 0, stream>>>(
        counts, cursor, tile_expert, slot_token);
    count_kernel<<<(T_TOK * TOPK + 255) / 256, 256, 0, stream>>>(topk_ids, counts);
    build_meta_kernel<<<1, 1, 0, stream>>>(counts, offs, tile_expert, tile_slot0);
    scatter_kernel<<<(T_TOK * TOPK + 255) / 256, 256, 0, stream>>>(
        topk_ids, topk_w, offs, cursor, slot_token, slot_weight);
    convert_x_kernel<<<(unsigned)((nout + 255) / 256), 256, 0, stream>>>(x, xb, nout);

    gemm1_kernel<<<dim3(RTILES, IDIM / NTILE), 256, 0, stream>>>(
        xb, w13p, w13s, w13z, tile_expert, tile_slot0, slot_token, hbuf);
    gemm2_kernel<<<dim3(RTILES, DDIM / NTILE), 256, 0, stream>>>(
        hbuf, w2p, w2s, w2z, tile_expert, tile_slot0, slot_token, slot_weight, out);
}
